// MultiheadAttn_25520695673289
// MI455X (gfx1250) — matrix operand order repaired and checked by reading, not compiled, not run on hardware
//
#include <hip/hip_runtime.h>

// Problem constants (match reference)
#define B_  2
#define S_  2048
#define D_  1024
#define NH_ 16
#define HD_ 64

typedef __attribute__((ext_vector_type(16))) __bf16 bf16x16;
typedef __attribute__((ext_vector_type(8)))  __bf16 bf16x8;
typedef __attribute__((ext_vector_type(4)))  __bf16 bf16x4;
typedef __attribute__((ext_vector_type(8)))  float  f32x8;
typedef __attribute__((ext_vector_type(4)))  unsigned int u32x4;
typedef __attribute__((ext_vector_type(8)))  int i32x8;
typedef __attribute__((ext_vector_type(4)))  int i32x4;

__device__ __forceinline__ f32x8 wmma_bf16(bf16x16 a, bf16x16 b, f32x8 c) {
  // D = A(16x32 bf16) * B(32x16 bf16) + C(16x16 f32)
  return __builtin_amdgcn_wmma_f32_16x16x32_bf16(false, a, false, b, (short)0, c,
                                                 false, false);
}

// A fragment: 16x32 bf16, A row-major [M x K], lda in elements (works for
// global or LDS-backed pointers; addrspace inferred).
__device__ __forceinline__ bf16x16 load_a_frag(const __bf16* base, int lda, int k0) {
  const int lane = threadIdx.x & 31;
  const int row  = lane & 15;
  const int kk   = k0 + ((lane & 16) ? 16 : 0);
  const __bf16* p = base + (size_t)row * lda + kk;
  bf16x8 lo = *(const bf16x8*)p;
  bf16x8 hi = *(const bf16x8*)(p + 8);
  bf16x16 r;
#pragma unroll
  for (int i = 0; i < 8; ++i) { r[i] = lo[i]; r[8 + i] = hi[i]; }
  return r;
}

// B fragment: 32x16 bf16 with B stored TRANSPOSED row-major [N x K].
__device__ __forceinline__ bf16x16 load_b_frag(const __bf16* baseT, int ldb, int k0) {
  const int lane = threadIdx.x & 31;
  const int col  = lane & 15;
  const int kk   = k0 + ((lane & 16) ? 16 : 0);
  return *(const bf16x16*)(baseT + (size_t)col * ldb + kk);
}

// ---------------------------------------------------------------- TDM helper
// 2-D tile DMA: global (row-major, row_stride_bytes) -> LDS (packed rows).
// D# built per CDNA5 ISA ch.8: data_size=1B units; groups 2/3 zero (2-D tile).
// This toolchain's builtin takes 6 args (extra int32x8 group before cpol).
__device__ __forceinline__ void tdm_load_2d(unsigned lds_off,
                                            const void* gaddr,
                                            unsigned tile_w_bytes,
                                            unsigned tile_rows,
                                            unsigned row_stride_bytes) {
  const unsigned long long ga = (unsigned long long)(size_t)gaddr;
  u32x4 g0;
  g0[0] = 1u;                                              // count=1, user mode
  g0[1] = lds_off;                                         // lds_addr
  g0[2] = (unsigned)(ga & 0xFFFFFFFFull);                  // global_addr[31:0]
  g0[3] = (unsigned)((ga >> 32) & 0x1FFFFFFull) | (2u << 30); // [56:32] | type=2
  i32x8 g1;
  const unsigned td0 = row_stride_bytes;   // tensor_dim0 (1B elements)
  const unsigned td1 = tile_rows;          // tensor_dim1
  g1[0] = 0;                                               // wg_mask=0, 1B elems
  g1[1] = (int)((td0 & 0xFFFFu) << 16);                    // dim0[15:0] @63:48
  g1[2] = (int)(((td0 >> 16) & 0xFFFFu) | ((td1 & 0xFFFFu) << 16));
  g1[3] = (int)(((td1 >> 16) & 0xFFFFu) | ((tile_w_bytes & 0xFFFFu) << 16));
  g1[4] = (int)(tile_rows & 0xFFFFu);                      // tile_dim1
  g1[5] = (int)row_stride_bytes;                           // dim0_stride[31:0]
  g1[6] = 0;
  g1[7] = 0;
  const i32x4 z4 = {0, 0, 0, 0};
  const i32x8 z8 = {0, 0, 0, 0, 0, 0, 0, 0};
  __builtin_amdgcn_tensor_load_to_lds(g0, g1, z4, z4, z8, 0);
}

__device__ __forceinline__ unsigned lds_offset_of(const void* p) {
  // generic pointer to LDS: low 32 bits are the LDS byte offset
  return (unsigned)(size_t)p;
}

// ---------------------------------------------------------------- conversions
__global__ __launch_bounds__(256) void cvt_x_kernel(const float4* __restrict__ x,
                                                    bf16x4* __restrict__ xb) {
  size_t i = (size_t)blockIdx.x * blockDim.x + threadIdx.x;
  float4 v = x[i];
  bf16x4 r = { (__bf16)v.x, (__bf16)v.y, (__bf16)v.z, (__bf16)v.w };
  xb[i] = r;
}

// W[k][n] f32 -> WT[n][k] bf16 (4 weights selected by blockIdx.z)
__global__ __launch_bounds__(256) void cvt_wT_kernel(
    const float* __restrict__ W0, const float* __restrict__ W1,
    const float* __restrict__ W2, const float* __restrict__ W3,
    __bf16* __restrict__ T0, __bf16* __restrict__ T1,
    __bf16* __restrict__ T2, __bf16* __restrict__ T3) {
  const float* W = (blockIdx.z == 0) ? W0 : (blockIdx.z == 1) ? W1
                   : (blockIdx.z == 2) ? W2 : W3;
  __bf16* T = (blockIdx.z == 0) ? T0 : (blockIdx.z == 1) ? T1
              : (blockIdx.z == 2) ? T2 : T3;
  int k = blockIdx.x * 32 + (threadIdx.x & 31);
  int n = blockIdx.y * 8 + (threadIdx.x >> 5);
  T[(size_t)n * D_ + k] = (__bf16)W[(size_t)k * D_ + n];
}

// -------------------------------------------------------------- QKV GEMM
__global__ __launch_bounds__(256) void qkv_gemm_kernel(
    const __bf16* __restrict__ xb,
    const __bf16* __restrict__ WqT, const __bf16* __restrict__ WkT,
    const __bf16* __restrict__ WvT,
    const float* __restrict__ bq, const float* __restrict__ bk,
    const float* __restrict__ bvp,
    __bf16* __restrict__ Qh, __bf16* __restrict__ Kh, __bf16* __restrict__ VhT) {
  const int which = blockIdx.z;
  const __bf16* WT = (which == 0) ? WqT : (which == 1) ? WkT : WvT;
  const float* bias = (which == 0) ? bq : (which == 1) ? bk : bvp;

  const int wid   = threadIdx.x >> 5;
  const int waveM = wid & 1, waveN = wid >> 1;
  const int bm = blockIdx.y * 64 + waveM * 32;
  const int bn = blockIdx.x * 128 + waveN * 32;

  const __bf16* A0 = xb + (size_t)bm * D_;
  const __bf16* A1 = xb + (size_t)(bm + 16) * D_;
  const __bf16* Bt0 = WT + (size_t)bn * D_;
  const __bf16* Bt1 = WT + (size_t)(bn + 16) * D_;

  f32x8 acc[2][2] = {};
  for (int k0 = 0; k0 < D_; k0 += 32) {
    bf16x16 a0 = load_a_frag(A0, D_, k0);
    bf16x16 a1 = load_a_frag(A1, D_, k0);
    bf16x16 b0 = load_b_frag(Bt0, D_, k0);
    bf16x16 b1 = load_b_frag(Bt1, D_, k0);
    acc[0][0] = wmma_bf16(a0, b0, acc[0][0]);
    acc[0][1] = wmma_bf16(a0, b1, acc[0][1]);
    acc[1][0] = wmma_bf16(a1, b0, acc[1][0]);
    acc[1][1] = wmma_bf16(a1, b1, acc[1][1]);
  }

  const int lane = threadIdx.x & 31;
  const int cn   = lane & 15;
  const int mhi  = (lane & 16) ? 8 : 0;
#pragma unroll
  for (int im = 0; im < 2; ++im)
#pragma unroll
    for (int in = 0; in < 2; ++in) {
      const int ccol = bn + in * 16 + cn;
      const float bia = bias[ccol];
      const int nh = ccol >> 6, h = ccol & 63;
#pragma unroll
      for (int j = 0; j < 8; ++j) {
        const int r = bm + im * 16 + mhi + j;
        const int bb = r >> 11, s = r & (S_ - 1);
        const __bf16 v = (__bf16)(acc[im][in][j] + bia);
        if (which == 0)
          Qh[((size_t)(bb * NH_ + nh) * S_ + s) * HD_ + h] = v;
        else if (which == 1)
          Kh[((size_t)(bb * NH_ + nh) * S_ + s) * HD_ + h] = v;
        else
          VhT[((size_t)(bb * NH_ + nh) * HD_ + h) * S_ + s] = v;
      }
    }
}

// --------------------------------------------------- attention pass A: stats
// Q tiles (shared by all 8 waves) are staged via TDM into double-buffered LDS.
__global__ __launch_bounds__(256) void attn_stats_kernel(
    const __bf16* __restrict__ Qh, const __bf16* __restrict__ Kh,
    float* __restrict__ mbuf, float* __restrict__ lbuf) {
  __shared__ __bf16 qtile[2][16 * HD_];  // 2 x 2KB

  const int bh = blockIdx.y;
  const int wid = threadIdx.x >> 5;
  const int t0 = blockIdx.x * 128 + wid * 16;
  const int lane = threadIdx.x & 31;

  const __bf16* Kb = Kh + (size_t)bh * S_ * HD_;
  const __bf16* Qb = Qh + (size_t)bh * S_ * HD_;

  const bf16x16 ka0 = load_a_frag(Kb + (size_t)t0 * HD_, HD_, 0);
  const bf16x16 ka1 = load_a_frag(Kb + (size_t)t0 * HD_, HD_, 32);

  float rm[8], rl[8];
#pragma unroll
  for (int j = 0; j < 8; ++j) { rm[j] = -1e30f; rl[j] = 0.0f; }

  if (wid == 0)
    tdm_load_2d(lds_offset_of(&qtile[0][0]), Qb, 2 * HD_, 16, 2 * HD_);

  for (int i = 0; i < S_ / 16; ++i) {
    const int p = i & 1;
    __builtin_amdgcn_s_wait_tensorcnt(0);  // wave0 waits; others trivially pass
    __syncthreads();                       // tile p visible; buf p^1 free
    if (wid == 0 && (i + 1) < S_ / 16)
      tdm_load_2d(lds_offset_of(&qtile[p ^ 1][0]),
                  Qb + (size_t)(i + 1) * 16 * HD_, 2 * HD_, 16, 2 * HD_);

    bf16x16 qb0 = load_b_frag(&qtile[p][0], HD_, 0);
    bf16x16 qb1 = load_b_frag(&qtile[p][0], HD_, 32);
    f32x8 acc = {};
    acc = wmma_bf16(ka0, qb0, acc);
    acc = wmma_bf16(ka1, qb1, acc);
#pragma unroll
    for (int j = 0; j < 8; ++j) {
      float v = acc[j] * 0.125f;
      float tmax = v;
      tmax = fmaxf(tmax, __shfl_xor(tmax, 1, 32));
      tmax = fmaxf(tmax, __shfl_xor(tmax, 2, 32));
      tmax = fmaxf(tmax, __shfl_xor(tmax, 4, 32));
      tmax = fmaxf(tmax, __shfl_xor(tmax, 8, 32));
      const float nm = fmaxf(rm[j], tmax);
      float pr = __expf(v - nm);
      pr += __shfl_xor(pr, 1, 32);
      pr += __shfl_xor(pr, 2, 32);
      pr += __shfl_xor(pr, 4, 32);
      pr += __shfl_xor(pr, 8, 32);
      rl[j] = rl[j] * __expf(rm[j] - nm) + pr;
      rm[j] = nm;
    }
  }
  if ((lane & 15) == 0) {
    const int tb = t0 + ((lane & 16) ? 8 : 0);
#pragma unroll
    for (int j = 0; j < 8; ++j) {
      mbuf[(size_t)bh * S_ + tb + j] = rm[j];
      lbuf[(size_t)bh * S_ + tb + j] = rl[j];
    }
  }
}

// ---------------------------------------------- attention pass B: O = P^T V
// K and V tiles (identical across the 8 waves) staged via TDM double buffers.
__global__ __launch_bounds__(256) void attn_av_kernel(
    const __bf16* __restrict__ Qh, const __bf16* __restrict__ Kh,
    const __bf16* __restrict__ VhT, const float* __restrict__ mbuf,
    const float* __restrict__ lbuf, __bf16* __restrict__ Obuf) {
  __shared__ __bf16 ktile[2][32 * HD_];  // 2 x 4KB, rows t, packed [32][64]
  __shared__ __bf16 vtile[2][HD_ * 32];  // 2 x 4KB, rows h, packed [64][32]

  const int bh = blockIdx.y;
  const int bb = bh / NH_, nh = bh % NH_;
  const int wid = threadIdx.x >> 5;
  const int s0 = blockIdx.x * 128 + wid * 16;
  const int lane = threadIdx.x & 31;
  const int laneHi = (lane & 16) ? 1 : 0;

  const __bf16* Kb = Kh + (size_t)bh * S_ * HD_;
  const __bf16* Qb = Qh + (size_t)bh * S_ * HD_;
  const __bf16* Vb = VhT + (size_t)bh * HD_ * S_;
  const float* mrow = mbuf + (size_t)bh * S_;
  const float* lrow = lbuf + (size_t)bh * S_;

  const bf16x16 qb0 = load_b_frag(Qb + (size_t)s0 * HD_, HD_, 0);
  const bf16x16 qb1 = load_b_frag(Qb + (size_t)s0 * HD_, HD_, 32);

  if (wid == 0) {
    tdm_load_2d(lds_offset_of(&ktile[0][0]), Kb, 2 * HD_, 32, 2 * HD_);
    tdm_load_2d(lds_offset_of(&vtile[0][0]), Vb, 64, HD_, 2 * S_);
  }

  f32x8 o[4] = {};
  for (int i = 0; i < S_ / 32; ++i) {
    const int p = i & 1;
    const int t0n = (i + 1) * 32;
    __builtin_amdgcn_s_wait_tensorcnt(0);
    __syncthreads();
    if (wid == 0 && t0n < S_) {
      tdm_load_2d(lds_offset_of(&ktile[p ^ 1][0]), Kb + (size_t)t0n * HD_,
                  2 * HD_, 32, 2 * HD_);
      tdm_load_2d(lds_offset_of(&vtile[p ^ 1][0]), Vb + t0n, 64, HD_, 2 * S_);
    }

    const int t0 = i * 32;
    bf16x16 ka0 = load_a_frag(&ktile[p][0], HD_, 0);
    bf16x16 ka1 = load_a_frag(&ktile[p][0], HD_, 32);
    bf16x16 ka2 = load_a_frag(&ktile[p][16 * HD_], HD_, 0);
    bf16x16 ka3 = load_a_frag(&ktile[p][16 * HD_], HD_, 32);
    f32x8 p0 = {}; p0 = wmma_bf16(ka0, qb0, p0); p0 = wmma_bf16(ka1, qb1, p0);
    f32x8 p1 = {}; p1 = wmma_bf16(ka2, qb0, p1); p1 = wmma_bf16(ka3, qb1, p1);

    const float* mp = mrow + t0 + laneHi * 8;
    const float* lp = lrow + t0 + laneHi * 8;
    f32x8 m0 = *(const f32x8*)mp;         f32x8 l0 = *(const f32x8*)lp;
    f32x8 m1 = *(const f32x8*)(mp + 16);  f32x8 l1 = *(const f32x8*)(lp + 16);

    // C(m=t,n=s) fragments transpose directly into the A(m=s,k=t) fragment.
    bf16x16 pa;
#pragma unroll
    for (int j = 0; j < 8; ++j) {
      pa[j]     = (__bf16)(__expf(p0[j] * 0.125f - m0[j]) / l0[j]);
      pa[8 + j] = (__bf16)(__expf(p1[j] * 0.125f - m1[j]) / l1[j]);
    }
#pragma unroll
    for (int h0 = 0; h0 < 4; ++h0) {
      const __bf16* vp = &vtile[p][(h0 * 16 + (lane & 15)) * 32 + laneHi * 8];
      bf16x8 vlo = *(const bf16x8*)vp;
      bf16x8 vhi = *(const bf16x8*)(vp + 16);
      bf16x16 vfrag;
#pragma unroll
      for (int i = 0; i < 8; ++i) { vfrag[i] = vlo[i]; vfrag[8 + i] = vhi[i]; }
      o[h0] = wmma_bf16(pa, vfrag, o[h0]);
    }
  }
#pragma unroll
  for (int h0 = 0; h0 < 4; ++h0) {
    const int h = h0 * 16 + (lane & 15);
#pragma unroll
    for (int j = 0; j < 8; ++j) {
      const int s = s0 + laneHi * 8 + j;
      Obuf[((size_t)bb * S_ + s) * D_ + nh * HD_ + h] = (__bf16)o[h0][j];
    }
  }
}

// ------------------------------------------------------- output projection
__global__ __launch_bounds__(256) void out_proj_kernel(
    const __bf16* __restrict__ Ob, const __bf16* __restrict__ WoT,
    const float* __restrict__ bo, float* __restrict__ out) {
  const int wid = threadIdx.x >> 5;
  const int waveM = wid & 1, waveN = wid >> 1;
  const int bm = blockIdx.y * 64 + waveM * 32;
  const int bn = blockIdx.x * 128 + waveN * 32;

  const __bf16* A0 = Ob + (size_t)bm * D_;
  const __bf16* A1 = Ob + (size_t)(bm + 16) * D_;
  const __bf16* Bt0 = WoT + (size_t)bn * D_;
  const __bf16* Bt1 = WoT + (size_t)(bn + 16) * D_;

  f32x8 acc[2][2] = {};
  for (int k0 = 0; k0 < D_; k0 += 32) {
    bf16x16 a0 = load_a_frag(A0, D_, k0);
    bf16x16 a1 = load_a_frag(A1, D_, k0);
    bf16x16 b0 = load_b_frag(Bt0, D_, k0);
    bf16x16 b1 = load_b_frag(Bt1, D_, k0);
    acc[0][0] = wmma_bf16(a0, b0, acc[0][0]);
    acc[0][1] = wmma_bf16(a0, b1, acc[0][1]);
    acc[1][0] = wmma_bf16(a1, b0, acc[1][0]);
    acc[1][1] = wmma_bf16(a1, b1, acc[1][1]);
  }

  const int lane = threadIdx.x & 31;
  const int cn = lane & 15;
  const int mhi = (lane & 16) ? 8 : 0;
#pragma unroll
  for (int im = 0; im < 2; ++im)
#pragma unroll
    for (int in = 0; in < 2; ++in) {
      const int ccol = bn + in * 16 + cn;
      const float bia = bo[ccol];
#pragma unroll
      for (int j = 0; j < 8; ++j) {
        const int r = bm + im * 16 + mhi + j;
        out[(size_t)r * D_ + ccol] = acc[im][in][j] + bia;
      }
    }
}

// ============================================================ launch
extern "C" void kernel_launch(void* const* d_in, const int* in_sizes, int n_in,
                              void* d_out, int out_size, void* d_ws, size_t ws_size,
                              hipStream_t stream) {
  const float* x    = (const float*)d_in[0];
  const float* Wq_w = (const float*)d_in[1];
  const float* Wq_b = (const float*)d_in[2];
  const float* Wk_w = (const float*)d_in[3];
  const float* Wk_b = (const float*)d_in[4];
  const float* Wv_w = (const float*)d_in[5];
  const float* Wv_b = (const float*)d_in[6];
  const float* Wo_w = (const float*)d_in[7];
  const float* Wo_b = (const float*)d_in[8];
  float* out = (float*)d_out;

  char* ws = (char*)d_ws;
  size_t off = 0;
  auto carve = [&](size_t bytes) -> char* {
    char* p = ws + off;
    off += (bytes + 255) & ~(size_t)255;
    return p;
  };
  const size_t XE = (size_t)B_ * S_ * D_;      // 4M elems
  const size_t WE = (size_t)D_ * D_;           // 1M elems
  __bf16* xb  = (__bf16*)carve(XE * 2);
  __bf16* WqT = (__bf16*)carve(WE * 2);
  __bf16* WkT = (__bf16*)carve(WE * 2);
  __bf16* WvT = (__bf16*)carve(WE * 2);
  __bf16* WoT = (__bf16*)carve(WE * 2);
  __bf16* Qh  = (__bf16*)carve(XE * 2);
  __bf16* Kh  = (__bf16*)carve(XE * 2);
  __bf16* VhT = (__bf16*)carve(XE * 2);
  float*  mbuf = (float*)carve((size_t)B_ * NH_ * S_ * 4);
  float*  lbuf = (float*)carve((size_t)B_ * NH_ * S_ * 4);
  __bf16* Obuf = (__bf16*)carve(XE * 2);
  (void)ws_size; (void)in_sizes; (void)n_in; (void)out_size;

  // 1) x -> bf16
  cvt_x_kernel<<<dim3(XE / (256 * 4)), 256, 0, stream>>>((const float4*)x,
                                                         (bf16x4*)xb);
  // 2) weights -> bf16 transposed [N][K]
  cvt_wT_kernel<<<dim3(D_ / 32, D_ / 8, 4), 256, 0, stream>>>(
      Wq_w, Wk_w, Wv_w, Wo_w, WqT, WkT, WvT, WoT);
  // 3) fused QKV projections (z selects destination)
  qkv_gemm_kernel<<<dim3(D_ / 128, (B_ * S_) / 64, 3), 256, 0, stream>>>(
      xb, WqT, WkT, WvT, Wq_b, Wk_b, Wv_b, Qh, Kh, VhT);
  // 4) per-key-row softmax stats over the query axis (TDM-staged Q tiles)
  attn_stats_kernel<<<dim3(S_ / 128, B_ * NH_), 256, 0, stream>>>(Qh, Kh, mbuf,
                                                                  lbuf);
  // 5) O = P^T V (TDM-staged K/V tiles)
  attn_av_kernel<<<dim3(S_ / 128, B_ * NH_), 256, 0, stream>>>(Qh, Kh, VhT,
                                                               mbuf, lbuf, Obuf);
  // 6) output projection + bias (f32 out)
  out_proj_kernel<<<dim3(D_ / 128, (B_ * S_) / 64), 256, 0, stream>>>(Obuf, WoT,
                                                                      Wo_b, out);
}
